// PerceiverEncoder_80676665688265
// MI455X (gfx1250) — matrix operand order repaired and checked by reading, not compiled, not run on hardware
//
#include <hip/hip_runtime.h>

typedef __attribute__((ext_vector_type(16))) __bf16 bf16x16;
typedef __attribute__((ext_vector_type(8)))  float  f32x8;
typedef __attribute__((ext_vector_type(4)))  int    i32x4;

__device__ __forceinline__ float gelu_tanh(float v) {
  const float c = 0.7978845608028654f;
  float u = c * (v + 0.044715f * v * v * v);
  return 0.5f * v * (1.0f + tanhf(u));
}

// CDNA5 async global->LDS copy (ASYNCcnt-tracked), 16 bytes per lane.
// LDS operand is the raw LDS offset = low 32 bits of the generic address.
__device__ __forceinline__ void async_copy_b128(const void* gaddr, void* lds) {
  unsigned           l = (unsigned)(unsigned long long)lds;
  unsigned long long g = (unsigned long long)gaddr;
  asm volatile("global_load_async_to_lds_b128 %0, %1, off"
               :: "v"(l), "v"(g) : "memory");
}
__device__ __forceinline__ void wait_async0() {
  asm volatile("s_wait_asynccnt 0x0" ::: "memory");
}

// ---------------------------------------------------------------------------
// Generic bf16 WMMA GEMM:  C[z] = act(alpha * A[z] @ B[z] + bias) (+ res[z])
//   z = blockIdx.z decomposed as (batch = z / Hdim, head = z % Hdim)
//   BTRANS=1 : B given as [N][K] row-major (pre-transposed weights / K^T)
//   BTRANS=0 : B given as [K][N] row-major (e.g. attn @ V)
//   Requires M%64==0, N%64==0, K%32==0 (true for every GEMM in this net).
// Block: 128 threads (4 waves, wave32); block tile 64x64, wave tile 32x32,
// K-step 32 -> v_wmma_f32_16x16x32_bf16.
// Double-buffered LDS: tile k0+32 is staged with global_load_async_to_lds_b128
// while tile k0 is consumed by the matrix pipe (global latency hidden).
// ---------------------------------------------------------------------------
template<int BTRANS, int ACT, int HASBIAS, int HASRES, int OUTBF16>
__global__ void __launch_bounds__(128)
gemm_wmma(const __bf16* __restrict__ A, long long sAb, long long sAh, int lda,
          const __bf16* __restrict__ Bm, long long sBb, long long sBh, int ldb,
          void* __restrict__ Cv, long long sCb, long long sCh, int ldc,
          const float* __restrict__ bias,
          const float* __restrict__ res, long long sRb, long long sRh, int ldr,
          int K, int Hdim, float alpha)
{
  __shared__ __attribute__((aligned(16))) __bf16 As[2][64][40];
  __shared__ __attribute__((aligned(16))) __bf16 Bs[2][64][40];
  const int tid  = threadIdx.x;
  const int wave = tid >> 5;
  const int lane = tid & 31;
  const int zb = blockIdx.z / Hdim;
  const int zh = blockIdx.z % Hdim;
  A  += (long long)zb * sAb + (long long)zh * sAh;
  Bm += (long long)zb * sBb + (long long)zh * sBh;
  const int rowBase = blockIdx.y * 64;
  const int colBase = blockIdx.x * 64;
  const int wm = (wave >> 1) * 32;
  const int wn = (wave & 1) * 32;

  // per-thread staging coordinates (2 x 16B chunks per tile per thread)
  const int r0 = tid >> 2,          kc0 = (tid & 3) << 3;          // chunk tid
  const int r1 = (tid + 128) >> 2,  kc1 = ((tid + 128) & 3) << 3;  // chunk tid+128

  auto stage = [&](int buf, int k0) {
    // A tile (64 rows x 32 k) via async global->LDS
    async_copy_b128(A + (long long)(rowBase + r0) * lda + (k0 + kc0), &As[buf][r0][kc0]);
    async_copy_b128(A + (long long)(rowBase + r1) * lda + (k0 + kc1), &As[buf][r1][kc1]);
    if (BTRANS) {
      async_copy_b128(Bm + (long long)(colBase + r0) * ldb + (k0 + kc0), &Bs[buf][r0][kc0]);
      async_copy_b128(Bm + (long long)(colBase + r1) * ldb + (k0 + kc1), &Bs[buf][r1][kc1]);
    } else {
      // B is [K][N]; transpose during LDS write (both loads stay in flight)
      int kr0 = tid >> 3,         nc0 = (tid & 7) << 3;
      int kr1 = (tid + 128) >> 3, nc1 = ((tid + 128) & 7) << 3;
      i32x4 t0 = *reinterpret_cast<const i32x4*>(
          Bm + (long long)(k0 + kr0) * ldb + (colBase + nc0));
      i32x4 t1 = *reinterpret_cast<const i32x4*>(
          Bm + (long long)(k0 + kr1) * ldb + (colBase + nc1));
      const __bf16* e0 = reinterpret_cast<const __bf16*>(&t0);
      const __bf16* e1 = reinterpret_cast<const __bf16*>(&t1);
      #pragma unroll
      for (int q = 0; q < 8; ++q) Bs[buf][nc0 + q][kr0] = e0[q];
      #pragma unroll
      for (int q = 0; q < 8; ++q) Bs[buf][nc1 + q][kr1] = e1[q];
    }
    // hint the tile after next (the one not yet in flight)
    if (k0 + 32 < K) {
      __builtin_prefetch(A + (long long)(rowBase + (tid >> 1)) * lda + (k0 + 32), 0, 1);
      if (BTRANS)
        __builtin_prefetch(Bm + (long long)(colBase + (tid >> 1)) * ldb + (k0 + 32), 0, 1);
      else
        __builtin_prefetch(Bm + (long long)(k0 + 32 + (tid & 31)) * ldb + colBase, 0, 1);
    }
  };

  f32x8 acc[2][2] = {};
  int cur = 0;
  stage(0, 0);

  for (int k0 = 0; k0 < K; k0 += 32) {
    wait_async0();          // this wave's writes into buffer `cur` are done
    __syncthreads();        // tile visible; everyone done reading buffer cur^1
    if (k0 + 32 < K) stage(cur ^ 1, k0 + 32);   // overlap with compute below

    // ---- build fragments per CDNA5 16-bit WMMA VGPR layouts ----
    const int mrow   = lane & 15;
    const int hiHalf = lane >> 4;      // 0 or 1
    const int kbaseA = hiHalf << 3;    // A: lanes 0-15 K{0..7,16..23}; 16-31 K{8..15,24..31}
    const int koffB  = hiHalf << 3;    // B: lanes 0-15 K{0..7,16..23}; 16-31 K{8..15,24..31}
    bf16x16 af[2], bfr[2];
    #pragma unroll
    for (int i = 0; i < 2; ++i) {
      union { bf16x16 v; i32x4 q[2]; } u;
      u.q[0] = *reinterpret_cast<const i32x4*>(&As[cur][wm + 16*i + mrow][kbaseA]);
      u.q[1] = *reinterpret_cast<const i32x4*>(&As[cur][wm + 16*i + mrow][kbaseA + 16]);
      af[i] = u.v;
    }
    #pragma unroll
    for (int j = 0; j < 2; ++j) {
      union { bf16x16 v; i32x4 q[2]; } u;
      u.q[0] = *reinterpret_cast<const i32x4*>(&Bs[cur][wn + 16*j + mrow][koffB]);
      u.q[1] = *reinterpret_cast<const i32x4*>(&Bs[cur][wn + 16*j + mrow][koffB + 16]);
      bfr[j] = u.v;
    }
    #pragma unroll
    for (int i = 0; i < 2; ++i)
      #pragma unroll
      for (int j = 0; j < 2; ++j)
        acc[i][j] = __builtin_amdgcn_wmma_f32_16x16x32_bf16(
            false, af[i], false, bfr[j], (short)0, acc[i][j], false, false);
    cur ^= 1;
  }

  // ---- epilogue: C layout VGPR r -> M = r + 8*(lane>=16), N = lane&15 ----
  const long long cb = (long long)zb * sCb + (long long)zh * sCh;
  const long long rb = (long long)zb * sRb + (long long)zh * sRh;
  const int mOff = (lane >> 4) << 3;
  const int nOff = lane & 15;
  #pragma unroll
  for (int i = 0; i < 2; ++i) {
    #pragma unroll
    for (int j = 0; j < 2; ++j) {
      #pragma unroll
      for (int r = 0; r < 8; ++r) {
        int grow = rowBase + wm + 16*i + r + mOff;
        int gcol = colBase + wn + 16*j + nOff;
        float v = acc[i][j][r] * alpha;
        if (HASBIAS) v += bias[gcol];
        if (ACT == 1) v = fmaxf(v, 0.0f);
        if (ACT == 2) v = gelu_tanh(v);
        if (HASRES) v += res[rb + (long long)grow * ldr + gcol];
        if (OUTBF16) ((__bf16*)Cv)[cb + (long long)grow * ldc + gcol] = (__bf16)v;
        else         ((float*)Cv)[cb + (long long)grow * ldc + gcol] = v;
      }
    }
  }
}

// ---------------------------------------------------------------------------
// LayerNorm over 512 columns, fp32 in -> bf16 out. One block (256 thr) / row.
// ---------------------------------------------------------------------------
__global__ void __launch_bounds__(256)
ln512(const float* __restrict__ x, const float* __restrict__ g,
      const float* __restrict__ b, __bf16* __restrict__ out)
{
  __shared__ float red[256];
  const int tid = threadIdx.x;
  const long long row = blockIdx.x;
  const float* xr = x + row * 512;
  float v0 = xr[tid], v1 = xr[tid + 256];
  red[tid] = v0 + v1;
  __syncthreads();
  for (int s = 128; s > 0; s >>= 1) {
    if (tid < s) red[tid] += red[tid + s];
    __syncthreads();
  }
  float mean = red[0] * (1.0f / 512.0f);
  __syncthreads();
  float d0 = v0 - mean, d1 = v1 - mean;
  red[tid] = d0 * d0 + d1 * d1;
  __syncthreads();
  for (int s = 128; s > 0; s >>= 1) {
    if (tid < s) red[tid] += red[tid + s];
    __syncthreads();
  }
  float inv = rsqrtf(red[0] * (1.0f / 512.0f) + 1e-5f);
  out[row * 512 + tid]       = (__bf16)(d0 * inv * g[tid] + b[tid]);
  out[row * 512 + tid + 256] = (__bf16)(d1 * inv * g[tid + 256] + b[tid + 256]);
}

// ---------------------------------------------------------------------------
// Row softmax: fp32 in -> bf16 out. One block (256 thr) / row.
// ---------------------------------------------------------------------------
__global__ void __launch_bounds__(256)
softmax_rows(const float* __restrict__ in, __bf16* __restrict__ out, int cols)
{
  __shared__ float red[256];
  const int tid = threadIdx.x;
  const long long row = blockIdx.x;
  const float* x = in + row * (long long)cols;
  float mx = -1e30f;
  for (int c = tid; c < cols; c += 256) mx = fmaxf(mx, x[c]);
  red[tid] = mx; __syncthreads();
  for (int s = 128; s > 0; s >>= 1) { if (tid < s) red[tid] = fmaxf(red[tid], red[tid + s]); __syncthreads(); }
  mx = red[0]; __syncthreads();
  float sum = 0.0f;
  for (int c = tid; c < cols; c += 256) sum += __expf(x[c] - mx);
  red[tid] = sum; __syncthreads();
  for (int s = 128; s > 0; s >>= 1) { if (tid < s) red[tid] += red[tid + s]; __syncthreads(); }
  float inv = 1.0f / red[0];
  __bf16* o = out + row * (long long)cols;
  for (int c = tid; c < cols; c += 256) o[c] = (__bf16)(__expf(x[c] - mx) * inv);
}

// ---------------------------------------------------------------------------
// NeRF positional encoding, padded K=48 -> 64 (zeros) so it feeds WMMA GEMM.
// enc[row, d*24 + s*12 + f] = (s ? cos : sin)(pos[row,d] * 2^(4f/11))
// ---------------------------------------------------------------------------
__global__ void __launch_bounds__(256)
nerf_enc_kernel(const float* __restrict__ pos, __bf16* __restrict__ enc, int total)
{
  int e = blockIdx.x * 256 + threadIdx.x;
  if (e >= total) return;
  int row = e >> 6, j = e & 63;
  float val = 0.0f;
  if (j < 48) {
    int d = j / 24, r = j % 24, f = r % 12, s = r / 12;
    float freq = exp2f(4.0f * (float)f / 11.0f);
    float a = pos[row * 2 + d] * freq;
    val = s ? __cosf(a) : __sinf(a);
  }
  enc[e] = (__bf16)val;
}

// vin = x @ W_in + b_in + kin   (K = 3, VALU)
__global__ void __launch_bounds__(256)
vin_kernel(const float* __restrict__ x, const float* __restrict__ W_in,
           const float* __restrict__ b_in, const float* __restrict__ kin,
           float* __restrict__ vin, int total)
{
  int e = blockIdx.x * 256 + threadIdx.x;
  if (e >= total) return;
  int row = e >> 9, c = e & 511;
  float v = b_in[c] + kin[e];
  v += x[row * 3 + 0] * W_in[c];
  v += x[row * 3 + 1] * W_in[512 + c];
  v += x[row * 3 + 2] * W_in[1024 + c];
  vin[e] = v;
}

// Wt[n][k] = bf16(W[k][n]) with K zero-padded to Kpad
__global__ void __launch_bounds__(256)
convT_kernel(const float* __restrict__ W, __bf16* __restrict__ Wt,
             int K, int N, int Kpad, int total)
{
  int e = blockIdx.x * 256 + threadIdx.x;
  if (e >= total) return;
  int n = e / Kpad, k = e % Kpad;
  Wt[e] = (k < K) ? (__bf16)W[(long long)k * N + n] : (__bf16)0.0f;
}

// ---------------------------------------------------------------------------
extern "C" void kernel_launch(void* const* d_in, const int* in_sizes, int n_in,
                              void* d_out, int out_size, void* d_ws, size_t ws_size,
                              hipStream_t stream)
{
  (void)in_sizes; (void)n_in; (void)out_size; (void)ws_size;
  constexpr int BATCH = 8, SEQ = 4096, HID = 512, NL = 256, DEPTH = 6;
  constexpr int HEADS = 8, FF = 2048;
  constexpr int ROWS = BATCH * SEQ;      // 32768
  constexpr int LROWS = BATCH * NL;      // 2048
  const float SCALE = 0.125f;            // 64^-0.5

  const float* x       = (const float*)d_in[0];
  const float* pos     = (const float*)d_in[1];
  const float* W_pos   = (const float*)d_in[2];
  const float* b_pos   = (const float*)d_in[3];
  const float* W_in    = (const float*)d_in[4];
  const float* b_in    = (const float*)d_in[5];
  const float* latents = (const float*)d_in[6];
  const float* lnq_g = (const float*)d_in[7],  *lnq_b = (const float*)d_in[8];
  const float* lnk_g = (const float*)d_in[9],  *lnk_b = (const float*)d_in[10];
  const float* lnv_g = (const float*)d_in[11], *lnv_b = (const float*)d_in[12];
  const float* cWq = (const float*)d_in[13], *cWk = (const float*)d_in[14];
  const float* cWv = (const float*)d_in[15], *cWo = (const float*)d_in[16];
  const float* c_bo = (const float*)d_in[17];
  const float* ln1_g = (const float*)d_in[18], *ln1_b = (const float*)d_in[19];
  const float* sWq = (const float*)d_in[20], *sWkv = (const float*)d_in[21];
  const float* sWo = (const float*)d_in[22], *s_bo = (const float*)d_in[23];
  const float* ln2_g = (const float*)d_in[24], *ln2_b = (const float*)d_in[25];
  const float* fW1 = (const float*)d_in[26], *f_b1 = (const float*)d_in[27];
  const float* fW2 = (const float*)d_in[28], *f_b2 = (const float*)d_in[29];
  const float* lnf_g = (const float*)d_in[30], *lnf_b = (const float*)d_in[31];
  const float* Wb = (const float*)d_in[32], *bbv = (const float*)d_in[33];

  char* base = (char*)d_ws;
  size_t cur = 0;
  auto alloc = [&](size_t bytes) -> size_t {
    size_t o = cur; cur += (bytes + 255) & ~(size_t)255; return o;
  };

  // bf16 transposed weights
  size_t o_wpos = alloc(512ull * 64 * 2);
  size_t o_cq = alloc(512ull * 512 * 2), o_ck = alloc(512ull * 512 * 2);
  size_t o_cv = alloc(512ull * 512 * 2), o_co = alloc(512ull * 512 * 2);
  size_t o_sq  = alloc(6ull * 512 * 512 * 2);
  size_t o_skv = alloc(6ull * 1024 * 512 * 2);
  size_t o_so  = alloc(6ull * 512 * 512 * 2);
  size_t o_f1  = alloc(6ull * 2048 * 512 * 2);
  size_t o_f2  = alloc(6ull * 512 * 2048 * 2);
  size_t o_wb  = alloc(64ull * 512 * 2);
  // activations
  size_t o_enc = alloc((size_t)ROWS * 64 * 2);
  size_t o_kin = alloc((size_t)ROWS * HID * 4);   // reused for cross-attn sim/attn
  size_t o_vin = alloc((size_t)ROWS * HID * 4);   // reused for latent-block scratch
  size_t o_kn  = alloc((size_t)ROWS * HID * 2);
  size_t o_vn  = alloc((size_t)ROWS * HID * 2);
  size_t o_qn  = alloc((size_t)NL * HID * 2);
  size_t o_qc  = alloc((size_t)NL * HID * 2);
  size_t o_kc  = alloc((size_t)ROWS * HID * 2);
  size_t o_vc  = alloc((size_t)ROWS * HID * 2);
  size_t o_lat = alloc((size_t)LROWS * HID * 4);
  size_t o_ob  = alloc((size_t)LROWS * HID * 2);
  // aliases (kin dead after kn+vin; vin dead after vn)
  size_t o_simb = o_kin;                                    // [H,256,4096] f32 per b
  size_t o_attb = o_kin + 8ull * 256 * 4096 * 4;            // [H,256,4096] bf16 per b
  size_t a = o_vin;
  size_t o_h1  = a; a += (size_t)LROWS * HID * 2;
  size_t o_qs  = a; a += (size_t)LROWS * HID * 2;
  size_t o_kvs = a; a += (size_t)LROWS * 1024 * 2;
  size_t o_ssim = a; a += 64ull * 256 * 256 * 4;
  size_t o_satt = a; a += 64ull * 256 * 256 * 2;
  size_t o_sob = a; a += (size_t)LROWS * HID * 2;
  size_t o_ff1 = a; a += (size_t)LROWS * FF * 2;
  size_t o_lnf = a; a += (size_t)LROWS * HID * 2;

  __bf16* wt_pos = (__bf16*)(base + o_wpos);
  __bf16* wt_cq = (__bf16*)(base + o_cq);   __bf16* wt_ck = (__bf16*)(base + o_ck);
  __bf16* wt_cv = (__bf16*)(base + o_cv);   __bf16* wt_co = (__bf16*)(base + o_co);
  __bf16* wt_sq = (__bf16*)(base + o_sq);   __bf16* wt_skv = (__bf16*)(base + o_skv);
  __bf16* wt_so = (__bf16*)(base + o_so);
  __bf16* wt_f1 = (__bf16*)(base + o_f1);   __bf16* wt_f2 = (__bf16*)(base + o_f2);
  __bf16* wt_b  = (__bf16*)(base + o_wb);
  __bf16* enc  = (__bf16*)(base + o_enc);
  float*  kinf = (float*)(base + o_kin);
  float*  vinf = (float*)(base + o_vin);
  __bf16* kn = (__bf16*)(base + o_kn);  __bf16* vn = (__bf16*)(base + o_vn);
  __bf16* qn = (__bf16*)(base + o_qn);
  __bf16* qc = (__bf16*)(base + o_qc);
  __bf16* kc = (__bf16*)(base + o_kc);  __bf16* vc = (__bf16*)(base + o_vc);
  float*  latf = (float*)(base + o_lat);
  __bf16* obuf = (__bf16*)(base + o_ob);
  float*  simb = (float*)(base + o_simb);
  __bf16* attb = (__bf16*)(base + o_attb);
  __bf16* h1  = (__bf16*)(base + o_h1);
  __bf16* qs  = (__bf16*)(base + o_qs);
  __bf16* kvs = (__bf16*)(base + o_kvs);
  float*  ssim = (float*)(base + o_ssim);
  __bf16* satt = (__bf16*)(base + o_satt);
  __bf16* sob = (__bf16*)(base + o_sob);
  __bf16* ff1 = (__bf16*)(base + o_ff1);
  __bf16* lnfb = (__bf16*)(base + o_lnf);
  float*  outp = (float*)d_out;

  auto convT = [&](const float* W, __bf16* Wt, int K, int Nn, int Kpad) {
    int total = Nn * Kpad;
    convT_kernel<<<(total + 255) / 256, 256, 0, stream>>>(W, Wt, K, Nn, Kpad, total);
  };

  // ---- weight conversion (fp32 -> bf16 transposed [N][K]) ----
  convT(W_pos, wt_pos, 48, 512, 64);
  convT(cWq, wt_cq, 512, 512, 512);
  convT(cWk, wt_ck, 512, 512, 512);
  convT(cWv, wt_cv, 512, 512, 512);
  convT(cWo, wt_co, 512, 512, 512);
  for (int i = 0; i < DEPTH; ++i) {
    convT(sWq  + (size_t)i * 512 * 512,  wt_sq  + (size_t)i * 512 * 512,  512, 512,  512);
    convT(sWkv + (size_t)i * 512 * 1024, wt_skv + (size_t)i * 1024 * 512, 512, 1024, 512);
    convT(sWo  + (size_t)i * 512 * 512,  wt_so  + (size_t)i * 512 * 512,  512, 512,  512);
    convT(fW1  + (size_t)i * 512 * 2048, wt_f1  + (size_t)i * 2048 * 512, 512, 2048, 512);
    convT(fW2  + (size_t)i * 2048 * 512, wt_f2  + (size_t)i * 512 * 2048, 2048, 512, 2048);
  }
  convT(Wb, wt_b, 512, 64, 512);

  // ---- input embeddings ----
  nerf_enc_kernel<<<(ROWS * 64) / 256, 256, 0, stream>>>(pos, enc, ROWS * 64);
  // kin = relu(enc @ W_pos + b_pos)
  gemm_wmma<1, 1, 1, 0, 0><<<dim3(512 / 64, ROWS / 64, 1), 128, 0, stream>>>(
      enc, 0, 0, 64, wt_pos, 0, 0, 64, (void*)kinf, 0, 0, 512,
      b_pos, nullptr, 0, 0, 0, 64, 1, 1.0f);
  vin_kernel<<<(ROWS * HID) / 256, 256, 0, stream>>>(x, W_in, b_in, kinf, vinf, ROWS * HID);

  // ---- cross-attention pre-norms + projections ----
  ln512<<<NL, 256, 0, stream>>>(latents, lnq_g, lnq_b, qn);
  ln512<<<ROWS, 256, 0, stream>>>(kinf, lnk_g, lnk_b, kn);
  ln512<<<ROWS, 256, 0, stream>>>(vinf, lnv_g, lnv_b, vn);
  gemm_wmma<1, 0, 0, 0, 1><<<dim3(8, NL / 64, 1), 128, 0, stream>>>(
      qn, 0, 0, 512, wt_cq, 0, 0, 512, (void*)qc, 0, 0, 512,
      nullptr, nullptr, 0, 0, 0, 512, 1, 1.0f);
  gemm_wmma<1, 0, 0, 0, 1><<<dim3(8, ROWS / 64, 1), 128, 0, stream>>>(
      kn, 0, 0, 512, wt_ck, 0, 0, 512, (void*)kc, 0, 0, 512,
      nullptr, nullptr, 0, 0, 0, 512, 1, 1.0f);
  gemm_wmma<1, 0, 0, 0, 1><<<dim3(8, ROWS / 64, 1), 128, 0, stream>>>(
      vn, 0, 0, 512, wt_cv, 0, 0, 512, (void*)vc, 0, 0, 512,
      nullptr, nullptr, 0, 0, 0, 512, 1, 1.0f);

  // ---- cross-attention, per batch element (reuses sim buffer) ----
  for (int b = 0; b < BATCH; ++b) {
    // sim[h] = SCALE * q[:,h] @ k_b[:,h]^T   -> [H,256,4096] f32
    gemm_wmma<1, 0, 0, 0, 0><<<dim3(SEQ / 64, NL / 64, HEADS), 128, 0, stream>>>(
        qc, 0, 64, 512,
        kc + (size_t)b * SEQ * HID, 0, 64, 512,
        (void*)simb, 0, 256ll * 4096, 4096,
        nullptr, nullptr, 0, 0, 0, 64, HEADS, SCALE);
    softmax_rows<<<HEADS * NL, 256, 0, stream>>>(simb, attb, SEQ);
    // o[:,h] = attn[h] @ v_b[:,h]   (B row-major [K,N])
    gemm_wmma<0, 0, 0, 0, 1><<<dim3(1, NL / 64, HEADS), 128, 0, stream>>>(
        attb, 0, 256ll * 4096, 4096,
        vc + (size_t)b * SEQ * HID, 0, 64, 512,
        (void*)(obuf + (size_t)b * NL * HID), 0, 64, 512,
        nullptr, nullptr, 0, 0, 0, SEQ, HEADS, 1.0f);
  }
  // lat = latents (broadcast) + obuf @ cWo + c_bo
  gemm_wmma<1, 0, 1, 1, 0><<<dim3(8, NL / 64, BATCH), 128, 0, stream>>>(
      obuf, 256ll * 512, 0, 512, wt_co, 0, 0, 512,
      (void*)latf, 256ll * 512, 0, 512, c_bo,
      latents, 0, 0, 512, 512, 1, 1.0f);

  // ---- latent self-attention + FFN blocks ----
  for (int i = 0; i < DEPTH; ++i) {
    ln512<<<LROWS, 256, 0, stream>>>(latf, ln1_g + i * 512, ln1_b + i * 512, h1);
    gemm_wmma<1, 0, 0, 0, 1><<<dim3(8, LROWS / 64, 1), 128, 0, stream>>>(
        h1, 0, 0, 512, wt_sq + (size_t)i * 512 * 512, 0, 0, 512,
        (void*)qs, 0, 0, 512, nullptr, nullptr, 0, 0, 0, 512, 1, 1.0f);
    gemm_wmma<1, 0, 0, 0, 1><<<dim3(16, LROWS / 64, 1), 128, 0, stream>>>(
        h1, 0, 0, 512, wt_skv + (size_t)i * 1024 * 512, 0, 0, 512,
        (void*)kvs, 0, 0, 1024, nullptr, nullptr, 0, 0, 0, 512, 1, 1.0f);
    // ssim[b,h] = SCALE * q[b,:,h] @ k[b,:,h]^T
    gemm_wmma<1, 0, 0, 0, 0><<<dim3(4, 4, BATCH * HEADS), 128, 0, stream>>>(
        qs, 256ll * 512, 64, 512,
        kvs, 256ll * 1024, 64, 1024,
        (void*)ssim, 8ll * 256 * 256, 256ll * 256, 256,
        nullptr, nullptr, 0, 0, 0, 64, HEADS, SCALE);
    softmax_rows<<<BATCH * HEADS * NL, 256, 0, stream>>>(ssim, satt, NL);
    // so[b,:,h] = attn[b,h] @ v[b,:,h]
    gemm_wmma<0, 0, 0, 0, 1><<<dim3(1, 4, BATCH * HEADS), 128, 0, stream>>>(
        satt, 8ll * 256 * 256, 256ll * 256, 256,
        kvs + 512, 256ll * 1024, 64, 1024,
        (void*)sob, 256ll * 512, 64, 512,
        nullptr, nullptr, 0, 0, 0, NL, HEADS, 1.0f);
    // lat += so @ sWo + s_bo
    gemm_wmma<1, 0, 1, 1, 0><<<dim3(8, LROWS / 64, 1), 128, 0, stream>>>(
        sob, 0, 0, 512, wt_so + (size_t)i * 512 * 512, 0, 0, 512,
        (void*)latf, 0, 0, 512, s_bo + i * 512,
        latf, 0, 0, 512, 512, 1, 1.0f);
    // FFN
    ln512<<<LROWS, 256, 0, stream>>>(latf, ln2_g + i * 512, ln2_b + i * 512, h1);
    gemm_wmma<1, 2, 1, 0, 1><<<dim3(FF / 64, LROWS / 64, 1), 128, 0, stream>>>(
        h1, 0, 0, 512, wt_f1 + (size_t)i * 2048 * 512, 0, 0, 512,
        (void*)ff1, 0, 0, FF, f_b1 + i * FF,
        nullptr, 0, 0, 0, 512, 1, 1.0f);
    gemm_wmma<1, 0, 1, 1, 0><<<dim3(8, LROWS / 64, 1), 128, 0, stream>>>(
        ff1, 0, 0, FF, wt_f2 + (size_t)i * 512 * 2048, 0, 0, 2048,
        (void*)latf, 0, 0, 512, f_b2 + i * 512,
        latf, 0, 0, 512, FF, 1, 1.0f);
  }

  // ---- final norm + bottleneck projection ----
  ln512<<<LROWS, 256, 0, stream>>>(latf, lnf_g, lnf_b, lnfb);
  gemm_wmma<1, 0, 1, 0, 0><<<dim3(1, LROWS / 64, 1), 128, 0, stream>>>(
      lnfb, 0, 0, 512, wt_b, 0, 0, 512, (void*)outp, 0, 0, 64,
      bbv, nullptr, 0, 0, 0, 512, 1, 1.0f);
}
